// GCNModel_23648089931788
// MI455X (gfx1250) — compile-verified, matrix operand order checked by reading, not run on hardware
//
#include <hip/hip_runtime.h>
#include <hip/hip_bf16.h>

// ---------------------------------------------------------------------------
// GCN forward on MI455X (gfx1250, wave32).
// GEMMs via v_wmma_f32_16x16x32_bf16 (f32 accum); edge propagation via
// coalesced wave-per-edge gather + global_atomic_add_f32 scatter.
// ---------------------------------------------------------------------------

typedef __bf16 bf16;
typedef __attribute__((ext_vector_type(8)))  __bf16 v8bf;
typedef __attribute__((ext_vector_type(16))) __bf16 v16bf;
typedef __attribute__((ext_vector_type(8)))  float  v8f;

__device__ __forceinline__ bf16 f2bf(float f) {
    // round-to-nearest-even f32 -> bf16
    unsigned u = __builtin_bit_cast(unsigned, f);
    u += 0x7FFFu + ((u >> 16) & 1u);
    unsigned short h = (unsigned short)(u >> 16);
    return __builtin_bit_cast(bf16, h);
}

__global__ void k_fill_f32(float* __restrict__ p, float v, int n) {
    int i = blockIdx.x * blockDim.x + threadIdx.x;
    if (i < n) p[i] = v;
}

__global__ void k_deg(const int* __restrict__ dst, float* __restrict__ deg, int E) {
    int e = blockIdx.x * blockDim.x + threadIdx.x;
    if (e < E) atomicAdd(&deg[dst[e]], 1.0f);
}

__global__ void k_rsqrt_inplace(float* __restrict__ p, int n) {
    int i = blockIdx.x * blockDim.x + threadIdx.x;
    if (i < n) p[i] = rsqrtf(p[i]);   // deg >= 1 always (self-loop)
}

__global__ void k_f32_to_bf16(const float* __restrict__ s, bf16* __restrict__ d, int n) {
    int i = blockIdx.x * blockDim.x + threadIdx.x;
    if (i < n) d[i] = f2bf(s[i]);
}

// W: [K][Ncols] row-major f32  ->  Wt: [Ncols][K] bf16 (N-major, contiguous K)
__global__ void k_transpose_bf16(const float* __restrict__ W, bf16* __restrict__ Wt,
                                 int K, int Ncols) {
    int i = blockIdx.x * blockDim.x + threadIdx.x;
    if (i < K * Ncols) {
        int k = i / Ncols, n = i - k * Ncols;
        Wt[n * K + k] = f2bf(W[i]);
    }
}

// C[M,Ncols] = A[M,128] @ B[128,Ncols], A row-major bf16, Bt = B^T (Ncols x 128) bf16.
// One wave per 16x16 tile; blockIdx.x -> row tile, blockIdx.y -> col tile.
// ISA 16-bit A/B fragment layout: lane l holds row/col (l&15); K-halves split by
// (l>>4): elements 0..7 = K[kb..kb+7], elements 8..15 = K[kb+16..kb+23], kb = kk + (l>>4)*8.
__global__ void k_gemm_bf16_wmma(const bf16* __restrict__ A, const bf16* __restrict__ Bt,
                                 float* __restrict__ C, int Ncols) {
    const int lane = threadIdx.x & 31;
    const int r    = lane & 15;
    const int half = lane >> 4;
    const int row0 = blockIdx.x * 16;
    const int col0 = blockIdx.y * 16;

    const bf16* arow = A  + (size_t)(row0 + r) * 128;
    const bf16* brow = Bt + (size_t)(col0 + r) * 128;

    v8f acc = {};
#pragma unroll
    for (int kk = 0; kk < 128; kk += 32) {
        const int kb = kk + half * 8;
        v8bf alo = *(const v8bf*)(arow + kb);
        v8bf ahi = *(const v8bf*)(arow + kb + 16);
        v8bf blo = *(const v8bf*)(brow + kb);
        v8bf bhi = *(const v8bf*)(brow + kb + 16);
        v16bf a, b;
#pragma unroll
        for (int i = 0; i < 8; ++i) {
            a[i] = alo[i]; a[i + 8] = ahi[i];
            b[i] = blo[i]; b[i + 8] = bhi[i];
        }
        acc = __builtin_amdgcn_wmma_f32_16x16x32_bf16(
            /*neg_a=*/false, a, /*neg_b=*/false, b,
            /*c_mod=*/(short)0, acc, /*reuse_a=*/false, /*reuse_b=*/false);
    }

    // C/D layout: lane l -> col (l&15); VGPR vr -> row vr + (l>>4)*8
    float* crow = C + (size_t)(row0 + half * 8) * Ncols + (col0 + r);
#pragma unroll
    for (int vr = 0; vr < 8; ++vr) crow[(size_t)vr * Ncols] = acc[vr];
}

// Wave-per-edge scatter: lanes cover `feat` features; coalesced 128B loads,
// global_atomic_add_f32 accumulation into agg[dst].
__global__ void k_scatter(const float* __restrict__ h, const int* __restrict__ src,
                          const int* __restrict__ dst, const float* __restrict__ dinv,
                          float* __restrict__ agg, int E, int feat) {
    int wid  = blockIdx.x * (blockDim.x >> 5) + (threadIdx.x >> 5);
    int lane = threadIdx.x & 31;
    if (wid >= E) return;
    int s = src[wid], d = dst[wid];
    float norm = dinv[s] * dinv[d];
    const float* hs = h + (size_t)s * feat;
    float*       ad = agg + (size_t)d * feat;
    for (int f = lane; f < feat; f += 32)
        atomicAdd(&ad[f], hs[f] * norm);
}

// out = agg + h*dinv^2 (self-loop folded in, atomic-free) + bias, optional relu.
__global__ void k_finalize(const float* __restrict__ agg, const float* __restrict__ h,
                           const float* __restrict__ dinv, const float* __restrict__ bias,
                           float* __restrict__ out, int n, int feat, int relu) {
    int i = blockIdx.x * blockDim.x + threadIdx.x;
    if (i >= n) return;
    int node = i / feat;
    int f    = i - node * feat;
    float di = dinv[node];
    float v  = agg[i] + h[i] * di * di + bias[f];
    out[i] = relu ? fmaxf(v, 0.0f) : v;
}

// Row-wise log_softmax over 64 logits: one wave per row, shuffle reductions.
__global__ void k_logsoftmax64(const float* __restrict__ logits, float* __restrict__ out, int N) {
    int row  = blockIdx.x * (blockDim.x >> 5) + (threadIdx.x >> 5);
    int lane = threadIdx.x & 31;
    if (row >= N) return;
    const float* lr = logits + (size_t)row * 64;
    float x0 = lr[lane], x1 = lr[lane + 32];
    float m = fmaxf(x0, x1);
#pragma unroll
    for (int off = 16; off > 0; off >>= 1) m = fmaxf(m, __shfl_xor(m, off, 32));
    float s = __expf(x0 - m) + __expf(x1 - m);
#pragma unroll
    for (int off = 16; off > 0; off >>= 1) s += __shfl_xor(s, off, 32);
    float lse = m + __logf(s);
    float* orow = out + (size_t)row * 64;
    orow[lane]      = x0 - lse;
    orow[lane + 32] = x1 - lse;
}

static inline int cdiv(int a, int b) { return (a + b - 1) / b; }

extern "C" void kernel_launch(void* const* d_in, const int* in_sizes, int n_in,
                              void* d_out, int out_size, void* d_ws, size_t ws_size,
                              hipStream_t stream) {
    const float* X  = (const float*)d_in[0];
    const int*   ei = (const int*)d_in[1];
    const float* W1 = (const float*)d_in[2];
    const float* b1 = (const float*)d_in[3];
    const float* W2 = (const float*)d_in[4];
    const float* b2 = (const float*)d_in[5];
    (void)n_in; (void)out_size; (void)ws_size;

    const int N = in_sizes[0] / 128;   // 50000 (multiple of 16)
    const int E = in_sizes[1] / 2;     // 800000
    const int* src = ei;
    const int* dst = ei + E;

    // ---- carve workspace (all offsets 256B-aligned) ----
    char* ws = (char*)d_ws;
    size_t off = 0;
    auto carve = [&](size_t bytes) -> void* {
        void* p = ws + off;
        off += (bytes + 255) & ~(size_t)255;
        return p;
    };
    float* dinv = (float*)carve((size_t)N * 4);            // deg -> dinv
    float* hbuf = (float*)carve((size_t)N * 128 * 4);      // GEMM outputs
    float* agg  = (float*)carve((size_t)N * 128 * 4);      // scatter accumulator / layer out
    bf16*  abf  = (bf16*)carve((size_t)N * 128 * 2);       // bf16 staging of GEMM input
    bf16*  wt1  = (bf16*)carve((size_t)128 * 128 * 2);     // W1^T bf16 [128n][128k]
    bf16*  wt2  = (bf16*)carve((size_t)64 * 128 * 2);      // W2^T bf16 [64n][128k]

    const int TB = 256;                 // 8 waves/block
    const int WPB = TB / 32;
    const dim3 g1(N / 16, 128 / 16);    // 3125 x 8 tiles
    const dim3 g2(N / 16, 64 / 16);     // 3125 x 4 tiles
    const int sgrid = cdiv(E, WPB);

    // ---- degrees: deg = 1 (self-loop) + count(dst); dinv = rsqrt(deg) ----
    k_fill_f32<<<cdiv(N, TB), TB, 0, stream>>>(dinv, 1.0f, N);
    k_deg<<<cdiv(E, TB), TB, 0, stream>>>(dst, dinv, E);
    k_rsqrt_inplace<<<cdiv(N, TB), TB, 0, stream>>>(dinv, N);

    // ---- weights -> transposed bf16 (one-time, tiny) ----
    k_transpose_bf16<<<cdiv(128 * 128, TB), TB, 0, stream>>>(W1, wt1, 128, 128);
    k_transpose_bf16<<<cdiv(128 * 64, TB), TB, 0, stream>>>(W2, wt2, 128, 64);

    // ---- layer 1: h = X@W1 ; h1 = agg + selfloop + b1 ----
    k_f32_to_bf16<<<cdiv(N * 128, TB), TB, 0, stream>>>(X, abf, N * 128);
    k_gemm_bf16_wmma<<<g1, 32, 0, stream>>>(abf, wt1, hbuf, 128);
    k_fill_f32<<<cdiv(N * 128, TB), TB, 0, stream>>>(agg, 0.0f, N * 128);
    k_scatter<<<sgrid, TB, 0, stream>>>(hbuf, src, dst, dinv, agg, E, 128);
    k_finalize<<<cdiv(N * 128, TB), TB, 0, stream>>>(agg, hbuf, dinv, b1, agg, N * 128, 128, 0);

    // ---- layer 2: t = h1@W1 ; h2 = relu(agg + selfloop + b1) ----
    k_f32_to_bf16<<<cdiv(N * 128, TB), TB, 0, stream>>>(agg, abf, N * 128);
    k_gemm_bf16_wmma<<<g1, 32, 0, stream>>>(abf, wt1, hbuf, 128);
    k_fill_f32<<<cdiv(N * 128, TB), TB, 0, stream>>>(agg, 0.0f, N * 128);
    k_scatter<<<sgrid, TB, 0, stream>>>(hbuf, src, dst, dinv, agg, E, 128);
    k_finalize<<<cdiv(N * 128, TB), TB, 0, stream>>>(agg, hbuf, dinv, b1, agg, N * 128, 128, 1);

    // ---- layer 3: u = h2@W2 ; logits = agg + selfloop + b2 ----
    k_f32_to_bf16<<<cdiv(N * 128, TB), TB, 0, stream>>>(agg, abf, N * 128);
    k_gemm_bf16_wmma<<<g2, 32, 0, stream>>>(abf, wt2, hbuf, 64);
    k_fill_f32<<<cdiv(N * 64, TB), TB, 0, stream>>>(agg, 0.0f, N * 64);
    k_scatter<<<sgrid, TB, 0, stream>>>(hbuf, src, dst, dinv, agg, E, 64);
    k_finalize<<<cdiv(N * 64, TB), TB, 0, stream>>>(agg, hbuf, dinv, b2, agg, N * 64, 64, 0);

    // ---- log_softmax over 64 classes -> d_out ----
    k_logsoftmax64<<<cdiv(N, WPB), TB, 0, stream>>>(agg, (float*)d_out, N);
}
